// TextEncoderV1_47682726920743
// MI455X (gfx1250) — compile-verified, matrix operand order checked by reading, not run on hardware
//
#include <hip/hip_runtime.h>
#include <hip/hip_bf16.h>
#include <cstddef>

#define DI __device__ __forceinline__

typedef __attribute__((ext_vector_type(16))) _Float16 v16h;
typedef __attribute__((ext_vector_type(8)))  _Float16 v8h;
typedef __attribute__((ext_vector_type(8)))  float    v8f;

// Model constants
#define H_   192
#define F_   768
#define NH_  2
#define KC_  96
#define WS_  10
#define OUT_ 192
#define B_   8
#define T_   1024
#define M_   (B_ * T_)   // 8192

// ---------------------------------------------------------------------------
// WMMA helpers (f16 -> f32 accum, 16x16x32).
// A tile (16x32, M x K): lane r=lane&15 holds row M=r; lanes<16 get K
// {k0..k0+7, k0+16..k0+23}, lanes>=16 get K {k0+8..k0+15, k0+24..k0+31}.
// B tile assumed mirror layout with N in place of M, so we feed it rows of
// W^T (i.e. weights in their native (O,C) row-major form).
// ---------------------------------------------------------------------------
DI v16h ld_tile(const _Float16* base, int ld, int lane) {
  const int r  = lane & 15;
  const int ko = (lane >> 4) << 3;
  const _Float16* p = base + r * ld + ko;
  v8h lo = *(const v8h*)(p);
  v8h hi = *(const v8h*)(p + 16);
  v16h o;
#pragma unroll
  for (int i = 0; i < 8; ++i) { o[i] = lo[i]; o[i + 8] = hi[i]; }
  return o;
}

DI v8f wmma_f16(v16h a, v16h b, v8f c) {
  return __builtin_amdgcn_wmma_f32_16x16x32_f16(false, a, false, b, (short)0, c,
                                                false, false);
}

// D layout: lane L: n = L&15 ; VGPR r holds row m = r + 8*(L>>4).

// ---------------------------------------------------------------------------
// Weight preparation
// ---------------------------------------------------------------------------
__global__ void k_f32_to_f16(const float* __restrict__ s, _Float16* __restrict__ d, int n) {
  int i = blockIdx.x * 256 + threadIdx.x;
  if (i < n) d[i] = (_Float16)s[i];
}

// src (R x C) f32 -> dst (C x R) f16
__global__ void k_transpose_f16(const float* __restrict__ s, _Float16* __restrict__ d,
                                int R, int C) {
  int i = blockIdx.x * 256 + threadIdx.x;
  if (i < R * C) {
    int r = i / C, c = i % C;
    d[c * R + r] = (_Float16)s[i];
  }
}

// src (O, C, 3) f32 -> dst (3, O, C) f16
__global__ void k_extract_conv(const float* __restrict__ s, _Float16* __restrict__ d,
                               int O, int C) {
  int i = blockIdx.x * 256 + threadIdx.x;
  int n = O * C * 3;
  if (i < n) {
    int k = i / (O * C);
    int rem = i - k * O * C;
    int o = rem / C, c = rem % C;
    d[i] = (_Float16)s[(o * C + c) * 3 + k];
  }
}

// ---------------------------------------------------------------------------
// Embedding: X = leaky(sqrt(H) * (phone @ Wp + bp + emb_pitch[pitch])) * mask
// A = phone f16 (M x 256), B = Wp^T f16 (192 x 256).
// Each wave: 32x64 output tile; block = 4 waves = 128x64.
// ---------------------------------------------------------------------------
__global__ void k_embed(const _Float16* __restrict__ ph, const _Float16* __restrict__ wt,
                        const float* __restrict__ bias, const float* __restrict__ embp,
                        const int* __restrict__ pitch, const int* __restrict__ lengths,
                        float* __restrict__ Xo, _Float16* __restrict__ X16,
                        _Float16* __restrict__ X16m) {
  const int lane = threadIdx.x & 31, wave = threadIdx.x >> 5;
  const int m0 = blockIdx.x * 128 + wave * 32;
  const int n0 = blockIdx.y * 64;
  v8f acc[4][2] = {};
  for (int k0 = 0; k0 < 256; k0 += 32) {
    v16h a0 = ld_tile(ph + (size_t)m0 * 256 + k0, 256, lane);
    v16h a1 = ld_tile(ph + (size_t)(m0 + 16) * 256 + k0, 256, lane);
#pragma unroll
    for (int j = 0; j < 4; ++j) {
      v16h b = ld_tile(wt + (size_t)(n0 + j * 16) * 256 + k0, 256, lane);
      acc[j][0] = wmma_f16(a0, b, acc[j][0]);
      acc[j][1] = wmma_f16(a1, b, acc[j][1]);
    }
  }
  const int nn = lane & 15, mh = (lane >> 4) * 8;
#pragma unroll
  for (int j = 0; j < 4; ++j) {
    int n = n0 + j * 16 + nn;
    float bv = bias[n];
#pragma unroll
    for (int p = 0; p < 2; ++p) {
#pragma unroll
      for (int r = 0; r < 8; ++r) {
        int m = m0 + p * 16 + mh + r;
        int bi = m / T_, t = m % T_;
        float v = acc[j][p][r] + bv + embp[(size_t)pitch[m] * H_ + n];
        v *= 13.856406460551018f;                  // sqrt(192)
        v = (v >= 0.0f) ? v : 0.1f * v;            // leaky relu
        if (t >= lengths[bi]) v = 0.0f;            // x * x_mask
        size_t off = (size_t)m * H_ + n;
        Xo[off] = v;
        X16[off] = (_Float16)v;
        X16m[off] = (_Float16)v;
      }
    }
  }
}

// ---------------------------------------------------------------------------
// Generic f16 linear: C(MxN) = A(MxK) @ W^T (W is N x K row-major), + bias.
// flags: bit0 relu, bit1 zero rows with t>=len.
// Any of out16/out32/outT may be null. outT stores (B, N, T) transposed.
// Wave tile 32x64; block 128x64.
// ---------------------------------------------------------------------------
__global__ void k_linear(const _Float16* __restrict__ A, const _Float16* __restrict__ W,
                         const float* __restrict__ bias,
                         _Float16* __restrict__ out16, float* __restrict__ out32,
                         _Float16* __restrict__ outT,
                         int N, int K, const int* __restrict__ lengths, int flags) {
  const int lane = threadIdx.x & 31, wave = threadIdx.x >> 5;
  const int m0 = blockIdx.x * 128 + wave * 32;
  const int n0 = blockIdx.y * 64;
  v8f acc[4][2] = {};
  for (int k0 = 0; k0 < K; k0 += 32) {
    v16h a0 = ld_tile(A + (size_t)m0 * K + k0, K, lane);
    v16h a1 = ld_tile(A + (size_t)(m0 + 16) * K + k0, K, lane);
#pragma unroll
    for (int j = 0; j < 4; ++j) {
      v16h b = ld_tile(W + (size_t)(n0 + j * 16) * K + k0, K, lane);
      acc[j][0] = wmma_f16(a0, b, acc[j][0]);
      acc[j][1] = wmma_f16(a1, b, acc[j][1]);
    }
  }
  const int nn = lane & 15, mh = (lane >> 4) * 8;
#pragma unroll
  for (int j = 0; j < 4; ++j) {
    int n = n0 + j * 16 + nn;
    float bv = bias ? bias[n] : 0.0f;
#pragma unroll
    for (int p = 0; p < 2; ++p) {
#pragma unroll
      for (int r = 0; r < 8; ++r) {
        int m = m0 + p * 16 + mh + r;
        int bi = m / T_, t = m % T_;
        float v = acc[j][p][r] + bv;
        if (flags & 1) v = fmaxf(v, 0.0f);
        if ((flags & 2) && t >= lengths[bi]) v = 0.0f;
        if (out16) out16[(size_t)m * N + n] = (_Float16)v;
        if (out32) out32[(size_t)m * N + n] = v;
        if (outT)  outT[((size_t)bi * N + n) * T_ + t] = (_Float16)v;
      }
    }
  }
}

// ---------------------------------------------------------------------------
// Time-conv1d (kernel 3, pad 1) as 3 shifted GEMMs.
// A (M x C) f16 (pre-masked), W3 (3, O, C) f16. flags: bit0 relu, bit1 mask.
// Wave tile 32x64; block 128x64.
// ---------------------------------------------------------------------------
__global__ void k_conv1d(const _Float16* __restrict__ A, const _Float16* __restrict__ W3,
                         const float* __restrict__ bias,
                         _Float16* __restrict__ out16, float* __restrict__ out32,
                         int O, int C, const int* __restrict__ lengths, int flags) {
  const int lane = threadIdx.x & 31, wave = threadIdx.x >> 5;
  const int m0 = blockIdx.x * 128 + wave * 32;
  const int n0 = blockIdx.y * 64;
  const int ko = (lane >> 4) << 3;
  const int r = lane & 15;
  const int ma = m0 + r;           // A rows this lane reads (two halves)
  const int mb = m0 + 16 + r;
  const int ta = ma % T_, tb = mb % T_;
  v8f acc[4][2] = {};
#pragma unroll
  for (int k = 0; k < 3; ++k) {
    int dk = k - 1;
    bool va = (ta + dk >= 0) && (ta + dk < T_);
    bool vb = (tb + dk >= 0) && (tb + dk < T_);
    const _Float16* rowa = A + (size_t)(ma + dk) * C + ko;
    const _Float16* rowb = A + (size_t)(mb + dk) * C + ko;
    const _Float16* Wk = W3 + (size_t)k * O * C;
    for (int c0 = 0; c0 < C; c0 += 32) {
      v16h a0, a1;
      if (va) {
        v8h lo = *(const v8h*)(rowa + c0);
        v8h hi = *(const v8h*)(rowa + c0 + 16);
#pragma unroll
        for (int i = 0; i < 8; ++i) { a0[i] = lo[i]; a0[i + 8] = hi[i]; }
      } else {
#pragma unroll
        for (int i = 0; i < 16; ++i) a0[i] = (_Float16)0.0f;
      }
      if (vb) {
        v8h lo = *(const v8h*)(rowb + c0);
        v8h hi = *(const v8h*)(rowb + c0 + 16);
#pragma unroll
        for (int i = 0; i < 8; ++i) { a1[i] = lo[i]; a1[i + 8] = hi[i]; }
      } else {
#pragma unroll
        for (int i = 0; i < 16; ++i) a1[i] = (_Float16)0.0f;
      }
#pragma unroll
      for (int j = 0; j < 4; ++j) {
        v16h b = ld_tile(Wk + (size_t)(n0 + j * 16) * C + c0, C, lane);
        acc[j][0] = wmma_f16(a0, b, acc[j][0]);
        acc[j][1] = wmma_f16(a1, b, acc[j][1]);
      }
    }
  }
  const int nn = lane & 15, mh = (lane >> 4) * 8;
#pragma unroll
  for (int j = 0; j < 4; ++j) {
    int n = n0 + j * 16 + nn;
    float bv = bias[n];
#pragma unroll
    for (int p = 0; p < 2; ++p) {
#pragma unroll
      for (int rr = 0; rr < 8; ++rr) {
        int mm = m0 + p * 16 + mh + rr;
        int bi = mm / T_, tt = mm % T_;
        float v = acc[j][p][rr] + bv;
        if (flags & 1) v = fmaxf(v, 0.0f);
        if ((flags & 2) && tt >= lengths[bi]) v = 0.0f;
        if (out16) out16[(size_t)mm * O + n] = (_Float16)v;
        if (out32) out32[(size_t)mm * O + n] = v;
      }
    }
  }
}

// ---------------------------------------------------------------------------
// Attention: one block per (b, h, 16-query-row tile). 128 threads (4 waves).
// Scores S = scale * (Q K^T) + banded rel-K term, mask, softmax, O = P V +
// banded rel-V term. S (16x1024 f32) and P (16x1024 f16) staged in LDS.
// ---------------------------------------------------------------------------
__global__ void k_attn(const _Float16* __restrict__ Q, const _Float16* __restrict__ Kf,
                       const _Float16* __restrict__ Vt,
                       const float* __restrict__ relk, const float* __restrict__ relv,
                       const int* __restrict__ lengths, _Float16* __restrict__ outp,
                       float scale) {
  extern __shared__ char smemraw[];
  float* smem = (float*)smemraw;                       // [16][1024] scores
  _Float16* pmem = (_Float16*)(smem + 16 * T_);        // [16][1024] probs
  float* red = (float*)(pmem + 16 * T_);               // [128] + [16]

  const int bh = blockIdx.y;
  const int b = bh / NH_, h = bh % NH_;
  const int t0 = blockIdx.x * 16;
  const int lane = threadIdx.x & 31, wave = threadIdx.x >> 5, tid = threadIdx.x;
  const int len = lengths[b];

  const _Float16* Qb = Q + ((size_t)b * T_ + t0) * H_ + h * KC_;
  const _Float16* Kb = Kf + (size_t)b * T_ * H_ + h * KC_;
  const _Float16* Vb = Vt + ((size_t)b * H_ + h * KC_) * T_;

  // Preload the three Q A-tiles (K-dim 96 = 3 x 32)
  v16h qa[3];
#pragma unroll
  for (int d = 0; d < 3; ++d) qa[d] = ld_tile(Qb + d * 32, H_, lane);

  // ---- scores via WMMA ----
  for (int st = wave; st < T_ / 16; st += 4) {
    int s0 = st * 16;
    v8f acc = {};
#pragma unroll
    for (int d = 0; d < 3; ++d) {
      v16h kb = ld_tile(Kb + (size_t)s0 * H_ + d * 32, H_, lane);
      acc = wmma_f16(qa[d], kb, acc);
    }
    int nn = lane & 15, mh = (lane >> 4) * 8;
#pragma unroll
    for (int r = 0; r < 8; ++r)
      smem[(mh + r) * T_ + s0 + nn] = acc[r] * scale;
  }
  __syncthreads();

  // ---- relative-K band: scores[t, t+j-10] += scale * q[t] . relk[j] ----
  for (int idx = tid; idx < 16 * 21; idx += 128) {
    int row = idx / 21, j = idx % 21;
    int t = t0 + row, s = t + j - WS_;
    if (s >= 0 && s < T_) {
      const _Float16* qp = Qb + (size_t)row * H_;
      const float* rk = relk + j * KC_;
      float d = 0.0f;
      for (int c = 0; c < KC_; ++c) d += (float)qp[c] * rk[c];
      smem[row * T_ + s] += d * scale;
    }
  }
  __syncthreads();

  // ---- attn_mask: -10000 where query or key is past length ----
  for (int idx = tid; idx < 16 * T_; idx += 128) {
    int row = idx >> 10, s = idx & (T_ - 1);
    int t = t0 + row;
    if (s >= len || t >= len) smem[idx] = -10000.0f;
  }
  __syncthreads();

  // ---- softmax: 8 threads per row ----
  const int row = tid >> 3, jj = tid & 7;
  float mx = -3.0e38f;
  for (int s = jj; s < T_; s += 8) mx = fmaxf(mx, smem[row * T_ + s]);
  red[row * 8 + jj] = mx;
  __syncthreads();
  if (jj == 0) {
    float m2 = red[row * 8];
#pragma unroll
    for (int q = 1; q < 8; ++q) m2 = fmaxf(m2, red[row * 8 + q]);
    red[128 + row] = m2;
  }
  __syncthreads();
  float rmax = red[128 + row];
  float sm = 0.0f;
  for (int s = jj; s < T_; s += 8) {
    float e = __expf(smem[row * T_ + s] - rmax);
    smem[row * T_ + s] = e;
    sm += e;
  }
  __syncthreads();
  red[row * 8 + jj] = sm;
  __syncthreads();
  if (jj == 0) {
    float s2 = 0.0f;
#pragma unroll
    for (int q = 0; q < 8; ++q) s2 += red[row * 8 + q];
    red[128 + row] = 1.0f / s2;
  }
  __syncthreads();
  float rinv = red[128 + row];
  for (int s = jj; s < T_; s += 8)
    pmem[row * T_ + s] = (_Float16)(smem[row * T_ + s] * rinv);
  __syncthreads();

  // ---- O = P V via WMMA (+ banded rel-V) ----
  for (int nt = wave; nt < KC_ / 16; nt += 4) {   // 6 tiles of 16 cols
    int d0 = nt * 16;
    v8f acc = {};
    for (int s0 = 0; s0 < T_; s0 += 32) {
      v16h pa = ld_tile(pmem + s0, T_, lane);
      v16h vb = ld_tile(Vb + (size_t)d0 * T_ + s0, T_, lane);
      acc = wmma_f16(pa, vb, acc);
    }
    int nn = lane & 15, mh = (lane >> 4) * 8;
#pragma unroll
    for (int r = 0; r < 8; ++r) {
      int rr = mh + r;
      int t = t0 + rr;
      int d = d0 + nn;
      float add = 0.0f;
#pragma unroll
      for (int j = 0; j < 21; ++j) {
        int s = t + j - WS_;
        if (s >= 0 && s < T_)
          add += (float)pmem[rr * T_ + s] * relv[j * KC_ + d];
      }
      outp[((size_t)b * T_ + t) * H_ + h * KC_ + d] = (_Float16)(acc[r] + add);
    }
  }
}

// ---------------------------------------------------------------------------
// x = LayerNorm_channels(x + y); emit f32, f16, masked f16. One block / (b,t).
// ---------------------------------------------------------------------------
__global__ void k_add_ln(const float* __restrict__ x, const float* __restrict__ y,
                         const float* __restrict__ g, const float* __restrict__ be,
                         const int* __restrict__ lengths,
                         float* __restrict__ xo, _Float16* __restrict__ x16,
                         _Float16* __restrict__ x16m) {
  __shared__ float s1[64], s2[64];
  const int m = blockIdx.x, tid = threadIdx.x;
  const int b = m / T_, t = m % T_;
  const float* xr = x + (size_t)m * H_;
  const float* yr = y + (size_t)m * H_;
  float vals[3], p0 = 0.0f, p1 = 0.0f;
#pragma unroll
  for (int i = 0; i < 3; ++i) {
    float v = xr[tid + 64 * i] + yr[tid + 64 * i];
    vals[i] = v; p0 += v; p1 += v * v;
  }
  s1[tid] = p0; s2[tid] = p1;
  __syncthreads();
  for (int o = 32; o > 0; o >>= 1) {
    if (tid < o) { s1[tid] += s1[tid + o]; s2[tid] += s2[tid + o]; }
    __syncthreads();
  }
  float mean = s1[0] * (1.0f / H_);
  float var = s2[0] * (1.0f / H_) - mean * mean;
  float rs = rsqrtf(var + 1e-5f);
  bool on = t < lengths[b];
#pragma unroll
  for (int i = 0; i < 3; ++i) {
    int c = tid + 64 * i;
    float o = (vals[i] - mean) * rs * g[c] + be[c];
    size_t off = (size_t)m * H_ + c;
    xo[off] = o;
    x16[off] = (_Float16)o;
    x16m[off] = (_Float16)(on ? o : 0.0f);
  }
}

// ---------------------------------------------------------------------------
// Final projection: stats = (x_masked @ projW^T + b) * mask, stored channel-
// major as m (B,OUT,T) then logs (B,OUT,T). Wave tile 32x64.
// ---------------------------------------------------------------------------
__global__ void k_proj(const _Float16* __restrict__ A, const _Float16* __restrict__ W,
                       const float* __restrict__ bias, const int* __restrict__ lengths,
                       float* __restrict__ out) {
  const int lane = threadIdx.x & 31, wave = threadIdx.x >> 5;
  const int m0 = blockIdx.x * 128 + wave * 32;
  const int n0 = blockIdx.y * 64;
  v8f acc[4][2] = {};
  for (int k0 = 0; k0 < H_; k0 += 32) {
    v16h a0 = ld_tile(A + (size_t)m0 * H_ + k0, H_, lane);
    v16h a1 = ld_tile(A + (size_t)(m0 + 16) * H_ + k0, H_, lane);
#pragma unroll
    for (int j = 0; j < 4; ++j) {
      v16h b = ld_tile(W + (size_t)(n0 + j * 16) * H_ + k0, H_, lane);
      acc[j][0] = wmma_f16(a0, b, acc[j][0]);
      acc[j][1] = wmma_f16(a1, b, acc[j][1]);
    }
  }
  const int nn = lane & 15, mh = (lane >> 4) * 8;
  const size_t BOT = (size_t)B_ * OUT_ * T_;
#pragma unroll
  for (int j = 0; j < 4; ++j) {
    int n = n0 + j * 16 + nn;
    float bv = bias[n];
#pragma unroll
    for (int p = 0; p < 2; ++p) {
#pragma unroll
      for (int r = 0; r < 8; ++r) {
        int m = m0 + p * 16 + mh + r;
        int bi = m / T_, t = m % T_;
        float v = acc[j][p][r] + bv;
        if (t >= lengths[bi]) v = 0.0f;
        size_t off = (n < OUT_)
            ? ((size_t)bi * OUT_ * T_ + (size_t)n * T_ + t)
            : (BOT + (size_t)bi * OUT_ * T_ + (size_t)(n - OUT_) * T_ + t);
        out[off] = v;
      }
    }
  }
}

__global__ void k_mask(const int* __restrict__ lengths, float* __restrict__ out) {
  int i = blockIdx.x * 256 + threadIdx.x;
  if (i < B_ * T_) {
    int b = i / T_, t = i % T_;
    out[(size_t)2 * B_ * OUT_ * T_ + i] = (t < lengths[b]) ? 1.0f : 0.0f;
  }
}

// ---------------------------------------------------------------------------
// Host orchestration
// ---------------------------------------------------------------------------
extern "C" void kernel_launch(void* const* d_in, const int* in_sizes, int n_in,
                              void* d_out, int out_size, void* d_ws, size_t ws_size,
                              hipStream_t stream) {
  (void)in_sizes; (void)n_in; (void)out_size; (void)ws_size;
  const float* phone = (const float*)d_in[0];
  const int* pitch = (const int*)d_in[1];
  const int* lengths = (const int*)d_in[2];
  const float* emb_phone_w = (const float*)d_in[3];
  const float* emb_phone_b = (const float*)d_in[4];
  const float* emb_pitch = (const float*)d_in[5];
  const float* proj_w = (const float*)d_in[6];
  const float* proj_b = (const float*)d_in[7];
  float* out = (float*)d_out;

  // workspace bump allocator (256B aligned)
  char* w = (char*)d_ws;
  auto alloc = [&](size_t bytes) -> void* {
    void* p = (void*)w;
    w += (bytes + 255) & ~(size_t)255;
    return p;
  };

  _Float16* ph16 = (_Float16*)alloc((size_t)M_ * 256 * 2);
  _Float16* wtE = (_Float16*)alloc((size_t)H_ * 256 * 2);
  _Float16* wtP = (_Float16*)alloc((size_t)2 * OUT_ * H_ * 2);
  _Float16 *wq16[6], *wk16[6], *wv16[6], *wo16[6], *w1x[6], *w2x[6];
  for (int l = 0; l < 6; ++l) {
    wq16[l] = (_Float16*)alloc((size_t)H_ * H_ * 2);
    wk16[l] = (_Float16*)alloc((size_t)H_ * H_ * 2);
    wv16[l] = (_Float16*)alloc((size_t)H_ * H_ * 2);
    wo16[l] = (_Float16*)alloc((size_t)H_ * H_ * 2);
    w1x[l] = (_Float16*)alloc((size_t)3 * F_ * H_ * 2);
    w2x[l] = (_Float16*)alloc((size_t)3 * H_ * F_ * 2);
  }
  float* X32 = (float*)alloc((size_t)M_ * H_ * 4);
  float* Y32 = (float*)alloc((size_t)M_ * H_ * 4);
  _Float16* X16 = (_Float16*)alloc((size_t)M_ * H_ * 2);
  _Float16* X16m = (_Float16*)alloc((size_t)M_ * H_ * 2);
  _Float16* Q16 = (_Float16*)alloc((size_t)M_ * H_ * 2);
  _Float16* K16 = (_Float16*)alloc((size_t)M_ * H_ * 2);
  _Float16* Vt16 = (_Float16*)alloc((size_t)M_ * H_ * 2);
  _Float16* AO16 = (_Float16*)alloc((size_t)M_ * H_ * 2);
  _Float16* Mid16 = (_Float16*)alloc((size_t)M_ * F_ * 2);

  // ---- weight prep ----
  {
    int n = M_ * 256;
    k_f32_to_f16<<<(n + 255) / 256, 256, 0, stream>>>(phone, ph16, n);
  }
  k_transpose_f16<<<(256 * H_ + 255) / 256, 256, 0, stream>>>(emb_phone_w, wtE, 256, H_);
  {
    int n = 2 * OUT_ * H_;
    k_f32_to_f16<<<(n + 255) / 256, 256, 0, stream>>>(proj_w, wtP, n);
  }
  for (int l = 0; l < 6; ++l) {
    const int base = 8 + 18 * l;
    int n = H_ * H_;
    k_f32_to_f16<<<(n + 255) / 256, 256, 0, stream>>>((const float*)d_in[base + 0], wq16[l], n);
    k_f32_to_f16<<<(n + 255) / 256, 256, 0, stream>>>((const float*)d_in[base + 2], wk16[l], n);
    k_f32_to_f16<<<(n + 255) / 256, 256, 0, stream>>>((const float*)d_in[base + 4], wv16[l], n);
    k_f32_to_f16<<<(n + 255) / 256, 256, 0, stream>>>((const float*)d_in[base + 6], wo16[l], n);
    int n1 = 3 * F_ * H_;
    k_extract_conv<<<(n1 + 255) / 256, 256, 0, stream>>>((const float*)d_in[base + 12], w1x[l], F_, H_);
    k_extract_conv<<<(n1 + 255) / 256, 256, 0, stream>>>((const float*)d_in[base + 14], w2x[l], H_, F_);
  }

  // ---- embedding ----
  dim3 blk(128);
  k_embed<<<dim3(M_ / 128, H_ / 64), blk, 0, stream>>>(
      ph16, wtE, emb_phone_b, emb_pitch, pitch, lengths, X32, X16, X16m);

  const float scale = 0.10206207261596577f;  // 1/sqrt(96)
  const size_t attn_smem = (size_t)16 * T_ * 4 + (size_t)16 * T_ * 2 + 1024;

  for (int l = 0; l < 6; ++l) {
    const int base = 8 + 18 * l;
    const float* bq = (const float*)d_in[base + 1];
    const float* bk = (const float*)d_in[base + 3];
    const float* bv = (const float*)d_in[base + 5];
    const float* bo = (const float*)d_in[base + 7];
    const float* relk = (const float*)d_in[base + 8];
    const float* relv = (const float*)d_in[base + 9];
    const float* g1 = (const float*)d_in[base + 10];
    const float* b1 = (const float*)d_in[base + 11];
    const float* fb1 = (const float*)d_in[base + 13];
    const float* fb2 = (const float*)d_in[base + 15];
    const float* g2 = (const float*)d_in[base + 16];
    const float* b2 = (const float*)d_in[base + 17];

    // Q, K, V projections (V stored transposed (B,H,T))
    k_linear<<<dim3(M_ / 128, H_ / 64), blk, 0, stream>>>(
        X16, wq16[l], bq, Q16, nullptr, nullptr, H_, H_, lengths, 0);
    k_linear<<<dim3(M_ / 128, H_ / 64), blk, 0, stream>>>(
        X16, wk16[l], bk, K16, nullptr, nullptr, H_, H_, lengths, 0);
    k_linear<<<dim3(M_ / 128, H_ / 64), blk, 0, stream>>>(
        X16, wv16[l], bv, nullptr, nullptr, Vt16, H_, H_, lengths, 0);

    // attention
    k_attn<<<dim3(T_ / 16, B_ * NH_), blk, attn_smem, stream>>>(
        Q16, K16, Vt16, relk, relv, lengths, AO16, scale);

    // output projection -> Y32, then x = LN(x + y)
    k_linear<<<dim3(M_ / 128, H_ / 64), blk, 0, stream>>>(
        AO16, wo16[l], bo, nullptr, Y32, nullptr, H_, H_, lengths, 0);
    k_add_ln<<<M_, 64, 0, stream>>>(X32, Y32, g1, b1, lengths, X32, X16, X16m);

    // FFN: conv3 (relu, masked) -> conv3 (masked) -> x = LN(x + y)
    k_conv1d<<<dim3(M_ / 128, F_ / 64), blk, 0, stream>>>(
        X16m, w1x[l], fb1, Mid16, nullptr, F_, H_, lengths, /*relu|mask*/ 3);
    k_conv1d<<<dim3(M_ / 128, H_ / 64), blk, 0, stream>>>(
        Mid16, w2x[l], fb2, nullptr, Y32, H_, F_, lengths, /*mask*/ 2);
    k_add_ln<<<M_, 64, 0, stream>>>(X32, Y32, g2, b2, lengths, X32, X16, X16m);
  }

  // ---- final projection (uses masked x) + mask output ----
  k_proj<<<dim3(M_ / 128, (2 * OUT_) / 64), blk, 0, stream>>>(
      X16m, wtP, proj_b, lengths, out);
  k_mask<<<(B_ * T_ + 255) / 256, 256, 0, stream>>>(lengths, out);
}
